// LinearAttention_83021717832017
// MI455X (gfx1250) — matrix operand order repaired and checked by reading, not compiled, not run on hardware
//
#include <hip/hip_runtime.h>
#include <hip/hip_bf16.h>
#include <cstddef>

// Problem constants (match reference)
#define BB     8
#define NN     16384
#define CIN    256      // DIM
#define DD     512      // D
#define HH     64       // DIM_HEAD
#define SCALEF 0.125f   // 64^-0.5
#define TM     32       // tokens per workgroup tile
#define NT     (NN / TM)   // 512 token tiles per batch

// LDS padding (in bf16 elements) for bank-conflict-free b128 fragment loads
#define XPAD   (CIN + 8)   // 264: row stride 528B -> lane base bank = 4*mr
#define KPAD   40          // 80B stride -> 20*nc mod 64 hits all 64 banks once
#define QPAD   (DD + 8)    // 520: row stride 1040B -> lane base bank = 4*mr

typedef __attribute__((ext_vector_type(16))) __bf16 v16bf;
typedef __attribute__((ext_vector_type(8)))  __bf16 v8bf;   // 16 bytes = one b128
typedef __attribute__((ext_vector_type(8)))  float  v8f;

union F8 { v8f v; float f[8]; };

__device__ inline v8f wmma_bf16(v16bf a, v16bf b, v8f c) {
  return __builtin_amdgcn_wmma_f32_16x16x32_bf16(
      false, a, false, b, (short)0, c, false, false);
}

// A fragment (16x32 bf16, row-major, row stride ld halfs).
// Layout: lane half 'hi' holds K = hi*16 + {0..15},
// i.e. 32 contiguous bytes -> 2x ds_load_b128.
__device__ inline v16bf load_a_frag(const __bf16* src, int ld, int lane) {
  const int hi = (lane >> 4) & 1;
  const __bf16* row = src + (size_t)(lane & 15) * ld + hi * 16;
  union { v16bf v; v8bf h[2]; } f;
  f.h[0] = *(const v8bf*)(row);
  f.h[1] = *(const v8bf*)(row + 8);
  return f.v;
}

// B fragment (32x16) from a K-transposed tile Wt[col][k] with row stride ldk.
// Lane holds column n = lane&15, K = hi*16 + {0..15}: 32 contiguous bytes.
__device__ inline v16bf load_b_frag_kt(const __bf16* base, int ldk, int lane) {
  const __bf16* p = base + (size_t)(lane & 15) * ldk + (lane >> 4) * 16;
  union { v16bf v; v8bf h[2]; } f;
  f.h[0] = *(const v8bf*)(p);
  f.h[1] = *(const v8bf*)(p + 8);
  return f.v;
}

// B fragment from a pre-transposed bf16 global matrix T[n][k] (row stride DD).
__device__ inline v16bf load_b_frag_gt(const __bf16* base, int lane) {
  const __bf16* p = base + (size_t)(lane & 15) * DD + (lane >> 4) * 16;
  union { v16bf v; v8bf h[2]; } f;
  f.h[0] = *(const v8bf*)(p);
  f.h[1] = *(const v8bf*)(p + 8);
  return f.v;
}

__device__ inline void store_bf16x2(__bf16* dst, float a, float b) {
  union { __bf16 h[2]; unsigned u; } pk;
  pk.h[0] = (__bf16)a; pk.h[1] = (__bf16)b;
  *(unsigned*)dst = pk.u;   // 4B-aligned by construction (even col, even pads)
}

// ---------------------------------------------------------------------------
// Kernel 1: WpWo = Wp@Wo (f32), cvec = bp@Wo + bo, WoT[h][k] = bf16(Wo[k][h])
// grid: 257 blocks x 256 threads (tiny, once per call)
// ---------------------------------------------------------------------------
__global__ __launch_bounds__(256) void k_prep(
    const float* __restrict__ Wp, const float* __restrict__ Wo,
    const float* __restrict__ bp, const float* __restrict__ bo,
    float* __restrict__ WpWo, float* __restrict__ cvec,
    __bf16* __restrict__ WoT) {
  if (blockIdx.x < 128) {
    const int idx = blockIdx.x * 256 + threadIdx.x;   // 0..32767
    const int d = idx >> 6, h = idx & 63;
    float s = 0.f;
    for (int e = 0; e < DD; ++e) s += Wp[(size_t)d * DD + e] * Wo[(size_t)e * HH + h];
    WpWo[(size_t)d * HH + h] = s;
  } else if (blockIdx.x == 128) {
    if (threadIdx.x < HH) {
      const int h = threadIdx.x;
      float s = bo[h];
      for (int d = 0; d < DD; ++d) s += bp[d] * Wo[(size_t)d * HH + h];
      cvec[h] = s;
    }
  } else {
    const int idx = (blockIdx.x - 129) * 256 + threadIdx.x;  // 0..32767
    const int h = idx >> 9, k = idx & 511;
    WoT[idx] = (__bf16)Wo[(size_t)k * HH + h];
  }
}

// ---------------------------------------------------------------------------
// Kernel 2 (pass 1): q = x@Wq + bq ; s = (q.qw)*SCALE ;
// partials[b][t][d] = sum_{rows in tile} s*q[d]  (fixed order, no atomics)
// grid: (NT, BB) x 256 threads (8 waves). Wave w: rt = w>>2, cb = (w&3)*128.
// ---------------------------------------------------------------------------
__global__ __launch_bounds__(256) void k_pass1(
    const float* __restrict__ x,  const float* __restrict__ Wq,
    const float* __restrict__ bq, const float* __restrict__ qw,
    float* __restrict__ partials) {
  __shared__ __align__(16) __bf16 Xs[TM * XPAD];   // 16.5 KB
  __shared__ __align__(16) __bf16 Wt[DD * KPAD];   // 40 KB, K-transposed slab
  __shared__ float scp[4][TM];
  __shared__ float sLds[TM];
  __shared__ float awp[2][DD];

  const int b = blockIdx.y, t = blockIdx.x, tid = threadIdx.x;
  const int wave = tid >> 5, lane = tid & 31;
  const int hi = lane >> 4, nc = lane & 15;
  const int rt = wave >> 2;
  const int cb = (wave & 3) * 128;

  const float* xb = x + ((size_t)b * NN + (size_t)t * TM) * CIN;
  for (int p = tid; p < (TM * CIN) / 2; p += 256) {
    const int row = p >> 7, col = (p & 127) * 2;
    const float2 v = *(const float2*)(xb + (size_t)row * CIN + col);
    store_bf16x2(&Xs[(size_t)row * XPAD + col], v.x, v.y);
  }

  F8 acc[8];
#pragma unroll
  for (int j = 0; j < 8; ++j)
#pragma unroll
    for (int r = 0; r < 8; ++r) acc[j].f[r] = 0.f;

  for (int ks = 0; ks < CIN / 32; ++ks) {
    __syncthreads();                          // prev slab consumed; Xs ready
    const float* wsrc = Wq + (size_t)ks * 32 * DD;
    for (int p = tid; p < (32 * DD) / 2; p += 256) {  // transpose-stage
      const int d = p & 511, k = (p >> 9) * 2;
      store_bf16x2(&Wt[(size_t)d * KPAD + k],
                   wsrc[(size_t)k * DD + d], wsrc[(size_t)(k + 1) * DD + d]);
    }
    __syncthreads();
    const v16bf a = load_a_frag(Xs + (size_t)(rt * 16) * XPAD + ks * 32, XPAD, lane);
#pragma unroll
    for (int j = 0; j < 8; ++j) {
      const v16bf bf = load_b_frag_kt(Wt + (size_t)(cb + j * 16) * KPAD, KPAD, lane);
      acc[j].v = wmma_bf16(a, bf, acc[j].v);
    }
  }

  // bias + per-row score partials
  float sacc[8];
#pragma unroll
  for (int r = 0; r < 8; ++r) sacc[r] = 0.f;
#pragma unroll
  for (int j = 0; j < 8; ++j) {
    const int col = cb + j * 16 + nc;
    const float bqv = bq[col], qwv = qw[col];
#pragma unroll
    for (int r = 0; r < 8; ++r) {
      acc[j].f[r] += bqv;
      sacc[r] += acc[j].f[r] * qwv;
    }
  }
#pragma unroll
  for (int m = 1; m <= 8; m <<= 1)
#pragma unroll
    for (int r = 0; r < 8; ++r) sacc[r] += __shfl_xor(sacc[r], m, 32);
  if (nc == 0) {
#pragma unroll
    for (int r = 0; r < 8; ++r)
      scp[wave & 3][rt * 16 + hi * 8 + r] = sacc[r];
  }
  __syncthreads();
  if (tid < TM)
    sLds[tid] = SCALEF * (scp[0][tid] + scp[1][tid] + scp[2][tid] + scp[3][tid]);
  __syncthreads();

  // aw partial per column
#pragma unroll
  for (int j = 0; j < 8; ++j) {
    float a = 0.f;
#pragma unroll
    for (int r = 0; r < 8; ++r)
      a += sLds[rt * 16 + hi * 8 + r] * acc[j].f[r];
    a += __shfl_xor(a, 16, 32);
    if (hi == 0) awp[rt][cb + j * 16 + nc] = a;
  }
  __syncthreads();
  float* pout = partials + ((size_t)b * NT + t) * DD;
  for (int d = tid; d < DD; d += 256) pout[d] = awp[0][d] + awp[1][d];
}

// ---------------------------------------------------------------------------
// Kernel 3: aw[d] = sum_t partials ; WcT[b][h][k] = bf16(aw[k] * WpWo[k][h])
// grid: (BB) x 256 threads, fixed-order sums.
// ---------------------------------------------------------------------------
__global__ __launch_bounds__(256) void k_reduce(
    const float* __restrict__ partials, const float* __restrict__ WpWo,
    __bf16* __restrict__ WcT) {
  __shared__ float awS[DD];
  const int b = blockIdx.x, tid = threadIdx.x;
  for (int d = tid; d < DD; d += 256) {
    const float* p = partials + (size_t)b * NT * DD + d;
    float s = 0.f;
    for (int t = 0; t < NT; ++t) s += p[(size_t)t * DD];
    awS[d] = s;
  }
  __syncthreads();
  __bf16* wc = WcT + (size_t)b * HH * DD;
  for (int i = tid; i < HH * DD; i += 256) {
    const int h = i >> 9, k = i & 511;
    wc[i] = (__bf16)(awS[k] * WpWo[(size_t)k * HH + h]);
  }
}

// ---------------------------------------------------------------------------
// Kernel 4 (pass 2): recompute q,k tiles; out = k@Wc_b + q@Wo + cvec.
// The slab buffer is reused to stage biased q (then k) as padded bf16
// A-operands for the two 512->64 contractions (one 16x16 out tile per wave).
// ---------------------------------------------------------------------------
__global__ __launch_bounds__(256) void k_pass2(
    const float* __restrict__ x,  const float* __restrict__ Wq,
    const float* __restrict__ bq, const float* __restrict__ Wk,
    const float* __restrict__ bk, const __bf16* __restrict__ WoT,
    const __bf16* __restrict__ WcT, const float* __restrict__ cvec,
    float* __restrict__ out) {
  __shared__ __align__(16) __bf16 Xs[TM * XPAD];     // 16.5 KB
  __shared__ __align__(16) __bf16 WsBuf[DD * KPAD];  // 40 KB: Wt slab / Qs,Ks

  const int b = blockIdx.y, t = blockIdx.x, tid = threadIdx.x;
  const int wave = tid >> 5, lane = tid & 31;
  const int hi = lane >> 4, nc = lane & 15;
  const int rt = wave >> 2;               // big-GEMM row tile (also out tile)
  const int cb = (wave & 3) * 128;        // big-GEMM col base
  const int c2 = (wave & 3) * 16;         // out-GEMM col base (H=64)

  const float* xb = x + ((size_t)b * NN + (size_t)t * TM) * CIN;
  for (int p = tid; p < (TM * CIN) / 2; p += 256) {
    const int row = p >> 7, col = (p & 127) * 2;
    const float2 v = *(const float2*)(xb + (size_t)row * CIN + col);
    store_bf16x2(&Xs[(size_t)row * XPAD + col], v.x, v.y);
  }

  F8 qacc[8], kacc[8];
#pragma unroll
  for (int j = 0; j < 8; ++j)
#pragma unroll
    for (int r = 0; r < 8; ++r) { qacc[j].f[r] = 0.f; kacc[j].f[r] = 0.f; }

  for (int ks = 0; ks < CIN / 32; ++ks) {
    __syncthreads();
    const float* wq = Wq + (size_t)ks * 32 * DD;
    for (int p = tid; p < (32 * DD) / 2; p += 256) {
      const int d = p & 511, k = (p >> 9) * 2;
      store_bf16x2(&WsBuf[(size_t)d * KPAD + k],
                   wq[(size_t)k * DD + d], wq[(size_t)(k + 1) * DD + d]);
    }
    __syncthreads();
    const v16bf a = load_a_frag(Xs + (size_t)(rt * 16) * XPAD + ks * 32, XPAD, lane);
#pragma unroll
    for (int j = 0; j < 8; ++j)
      qacc[j].v = wmma_bf16(a,
          load_b_frag_kt(WsBuf + (size_t)(cb + j * 16) * KPAD, KPAD, lane), qacc[j].v);
    __syncthreads();
    const float* wk = Wk + (size_t)ks * 32 * DD;
    for (int p = tid; p < (32 * DD) / 2; p += 256) {
      const int d = p & 511, k = (p >> 9) * 2;
      store_bf16x2(&WsBuf[(size_t)d * KPAD + k],
                   wk[(size_t)k * DD + d], wk[(size_t)(k + 1) * DD + d]);
    }
    __syncthreads();
#pragma unroll
    for (int j = 0; j < 8; ++j)
      kacc[j].v = wmma_bf16(a,
          load_b_frag_kt(WsBuf + (size_t)(cb + j * 16) * KPAD, KPAD, lane), kacc[j].v);
  }

  // biases
#pragma unroll
  for (int j = 0; j < 8; ++j) {
    const int col = cb + j * 16 + nc;
    const float bqv = bq[col], bkv = bk[col];
#pragma unroll
    for (int r = 0; r < 8; ++r) { qacc[j].f[r] += bqv; kacc[j].f[r] += bkv; }
  }

  // stage biased q (bf16, padded rows) and contract with WoT
  __syncthreads();
#pragma unroll
  for (int j = 0; j < 8; ++j)
#pragma unroll
    for (int r = 0; r < 8; ++r)
      WsBuf[(size_t)(rt * 16 + hi * 8 + r) * QPAD + cb + j * 16 + nc] =
          (__bf16)qacc[j].f[r];
  __syncthreads();

  F8 oacc;
#pragma unroll
  for (int r = 0; r < 8; ++r) oacc.f[r] = 0.f;
  for (int ks = 0; ks < DD / 32; ++ks) {
    const v16bf a2 = load_a_frag(WsBuf + (size_t)(rt * 16) * QPAD + ks * 32, QPAD, lane);
    const v16bf b2 = load_b_frag_gt(WoT + (size_t)c2 * DD + ks * 32, lane);
    oacc.v = wmma_bf16(a2, b2, oacc.v);
  }

  // stage biased k and contract with per-batch WcT
  __syncthreads();
#pragma unroll
  for (int j = 0; j < 8; ++j)
#pragma unroll
    for (int r = 0; r < 8; ++r)
      WsBuf[(size_t)(rt * 16 + hi * 8 + r) * QPAD + cb + j * 16 + nc] =
          (__bf16)kacc[j].f[r];
  __syncthreads();

  const __bf16* Wcb = WcT + (size_t)b * HH * DD;
  for (int ks = 0; ks < DD / 32; ++ks) {
    const v16bf a2 = load_a_frag(WsBuf + (size_t)(rt * 16) * QPAD + ks * 32, QPAD, lane);
    const v16bf b2 = load_b_frag_gt(Wcb + (size_t)c2 * DD + ks * 32, lane);
    oacc.v = wmma_bf16(a2, b2, oacc.v);
  }

  const float cv = cvec[c2 + nc];
  float* ob = out + ((size_t)b * NN + (size_t)t * TM) * HH;
#pragma unroll
  for (int r = 0; r < 8; ++r)
    ob[(size_t)(rt * 16 + hi * 8 + r) * HH + c2 + nc] = oacc.f[r] + cv;
}

// ---------------------------------------------------------------------------
extern "C" void kernel_launch(void* const* d_in, const int* in_sizes, int n_in,
                              void* d_out, int out_size, void* d_ws, size_t ws_size,
                              hipStream_t stream) {
  (void)in_sizes; (void)n_in; (void)out_size; (void)ws_size;
  const float* x  = (const float*)d_in[0];
  const float* Wq = (const float*)d_in[1];
  const float* bq = (const float*)d_in[2];
  const float* Wk = (const float*)d_in[3];
  const float* bk = (const float*)d_in[4];
  const float* qw = (const float*)d_in[5];
  const float* Wp = (const float*)d_in[6];
  const float* bp = (const float*)d_in[7];
  const float* Wo = (const float*)d_in[8];
  const float* bo = (const float*)d_in[9];
  float* out = (float*)d_out;

  // workspace layout: f32 {WpWo | cvec | partials} then bf16 {WoT | WcT}
  float* wsf      = (float*)d_ws;
  float* WpWo     = wsf;                               // 512*64 f32
  float* cvec     = WpWo + (size_t)DD * HH;            // 64 f32
  float* partials = cvec + HH;                         // 8*512*512 f32
  __bf16* WoT     = (__bf16*)(partials + (size_t)BB * NT * DD);  // 64*512 bf16
  __bf16* WcT     = WoT + (size_t)HH * DD;                       // 8*64*512 bf16

  k_prep<<<257, 256, 0, stream>>>(Wp, Wo, bp, bo, WpWo, cvec, WoT);
  k_pass1<<<dim3(NT, BB), 256, 0, stream>>>(x, Wq, bq, qw, partials);
  k_reduce<<<BB, 256, 0, stream>>>(partials, WpWo, WcT);
  k_pass2<<<dim3(NT, BB), 256, 0, stream>>>(x, Wq, bq, Wk, bk, WoT, WcT, cvec, out);
}
